// FullAttention_DS_integ_57810259804510
// MI455X (gfx1250) — matrix operand order repaired and checked by reading, not compiled, not run on hardware
//
#include <hip/hip_runtime.h>

// ---------------------------------------------------------------------------
// Attention over heads: for each (b,n): S = Q·K^T [128x128], softmax rows,
// O = A·V [128x64].  B=4, N=1024, H=J=128, E=D=64.
// One 256-thread block (8 wave32) per (b,n). f16 WMMA, fp32 accumulate.
// ---------------------------------------------------------------------------

typedef __attribute__((ext_vector_type(16))) _Float16 v16h;
typedef __attribute__((ext_vector_type(8)))  _Float16 v8h;
typedef __attribute__((ext_vector_type(4)))  _Float16 h4;
typedef __attribute__((ext_vector_type(8)))  float    v8f;

#define BATCH 4
#define NSEQ  1024
#define HDIM  128   // heads == J
#define EDIM  64    // E == D

__device__ __forceinline__ v16h cat8(v8h lo, v8h hi) {
  return __builtin_shufflevector(lo, hi, 0,1,2,3,4,5,6,7,8,9,10,11,12,13,14,15);
}

__global__ __launch_bounds__(256)
void attn_heads_wmma_kernel(const float* __restrict__ Q,
                            const float* __restrict__ K,
                            const float* __restrict__ V,
                            float* __restrict__ O)
{
  // LDS staging (f16): Q,K row-major [row][e], V transposed [d][j], P per-wave.
  __shared__ _Float16 Qh[HDIM * EDIM];        // 16 KB
  __shared__ _Float16 Kh[HDIM * EDIM];        // 16 KB
  __shared__ _Float16 Vt[EDIM * HDIM];        // 16 KB (transposed)
  __shared__ _Float16 Ph[8 * 16 * HDIM];      // 32 KB (per-wave P [16][128])

  const int tid  = threadIdx.x;
  const int bn   = blockIdx.x;                       // b*N + n
  const size_t base = (size_t)bn * HDIM * EDIM;      // 8192 floats per tile

  // ---- cooperative load: global fp32 -> LDS f16 (V transposed) ----
  const float4* Qg = (const float4*)(Q + base);
  const float4* Kg = (const float4*)(K + base);
  const float4* Vg = (const float4*)(V + base);
#pragma unroll
  for (int rep = 0; rep < 8; ++rep) {
    const int i4  = tid + rep * 256;     // float4 index, 2048 total
    const int row = i4 >> 4;             // 0..127
    const int col = (i4 & 15) << 2;      // 0..60 step 4
    float4 q = Qg[i4];
    float4 k = Kg[i4];
    float4 v = Vg[i4];
    *(h4*)(&Qh[row * EDIM + col]) =
        (h4){(_Float16)q.x, (_Float16)q.y, (_Float16)q.z, (_Float16)q.w};
    *(h4*)(&Kh[row * EDIM + col]) =
        (h4){(_Float16)k.x, (_Float16)k.y, (_Float16)k.z, (_Float16)k.w};
    Vt[(col + 0) * HDIM + row] = (_Float16)v.x;
    Vt[(col + 1) * HDIM + row] = (_Float16)v.y;
    Vt[(col + 2) * HDIM + row] = (_Float16)v.z;
    Vt[(col + 3) * HDIM + row] = (_Float16)v.w;
  }
  __syncthreads();

  const int wave  = tid >> 5;
  const int lane  = tid & 31;
  const int lhalf = lane >> 4;           // 0: lanes 0-15, 1: lanes 16-31
  const int l15   = lane & 15;
  const int m0    = wave * 16;           // this wave's 16 query rows

  // ---- A-operand (Q rows), two K-chunks of 32 ----
  v16h qa[2];
  {
    const _Float16* qrow = &Qh[(m0 + l15) * EDIM];
#pragma unroll
    for (int c = 0; c < 2; ++c) {
      const int k0 = c * 32 + lhalf * 8;             // A layout half-wave split
      v8h lo = *(const v8h*)(qrow + k0);
      v8h hi = *(const v8h*)(qrow + k0 + 16);
      qa[c] = cat8(lo, hi);
    }
  }

  // ---- scores: S[16 x 128] = Q_w · K^T, 8 j-tiles x 2 k-chunks ----
  v8f s[8];
#pragma unroll
  for (int jt = 0; jt < 8; ++jt) {
    const _Float16* krow = &Kh[(jt * 16 + l15) * EDIM];   // column N = l15
    v8f acc = {};
#pragma unroll
    for (int c = 0; c < 2; ++c) {
      const int k0 = c * 32 + lhalf * 8;             // B layout half-wave split
      v8h lo = *(const v8h*)(krow + k0);
      v8h hi = *(const v8h*)(krow + k0 + 16);
      v16h kb = cat8(lo, hi);
      acc = __builtin_amdgcn_wmma_f32_16x16x32_f16(
          false, qa[c], false, kb, (short)0, acc, false, false);
    }
    s[jt] = acc;
  }

  // ---- softmax over j (per row m = r + 8*lhalf, reduced across 16 lanes) ----
  const float sc = 0.125f * 1.44269504088896340736f;   // (1/sqrt(E)) * log2(e)
  float rsum[8];
#pragma unroll
  for (int r = 0; r < 8; ++r) {
    float mx = s[0][r];
#pragma unroll
    for (int jt = 1; jt < 8; ++jt) mx = fmaxf(mx, s[jt][r]);
#pragma unroll
    for (int off = 1; off < 16; off <<= 1)
      mx = fmaxf(mx, __shfl_xor(mx, off, 32));         // stays within 16-lane half
    float sum = 0.0f;
#pragma unroll
    for (int jt = 0; jt < 8; ++jt) {
      float p = __builtin_amdgcn_exp2f((s[jt][r] - mx) * sc);
      s[jt][r] = p;
      sum += p;
    }
#pragma unroll
    for (int off = 1; off < 16; off <<= 1)
      sum += __shfl_xor(sum, off, 32);
    rsum[r] = sum;
  }

  // ---- spill P (f16) to LDS to re-layout C-tile -> A-operand ----
  _Float16* Pw = &Ph[wave * 16 * HDIM];
#pragma unroll
  for (int jt = 0; jt < 8; ++jt) {
    const int j = jt * 16 + l15;
#pragma unroll
    for (int r = 0; r < 8; ++r) {
      const int m = r + 8 * lhalf;
      Pw[m * HDIM + j] = (_Float16)s[jt][r];
    }
  }
  __syncthreads();

  // ---- P A-operand chunks (4 chunks of K=32 over j) ----
  v16h pa[4];
  {
    const _Float16* prow = &Pw[l15 * HDIM];
#pragma unroll
    for (int c = 0; c < 4; ++c) {
      const int k0 = c * 32 + lhalf * 8;
      v8h lo = *(const v8h*)(prow + k0);
      v8h hi = *(const v8h*)(prow + k0 + 16);
      pa[c] = cat8(lo, hi);
    }
  }

  float rinv[8];
#pragma unroll
  for (int r = 0; r < 8; ++r) rinv[r] = 1.0f / rsum[r];

  // ---- O[16 x 64] = P · V, 4 d-tiles x 4 k-chunks; normalize and store ----
#pragma unroll
  for (int dt = 0; dt < 4; ++dt) {
    const _Float16* vrow = &Vt[(dt * 16 + l15) * HDIM];  // column N = l15 (d)
    v8f acc = {};
#pragma unroll
    for (int c = 0; c < 4; ++c) {
      const int k0 = c * 32 + lhalf * 8;
      v8h lo = *(const v8h*)(vrow + k0);
      v8h hi = *(const v8h*)(vrow + k0 + 16);
      v16h vb = cat8(lo, hi);
      acc = __builtin_amdgcn_wmma_f32_16x16x32_f16(
          false, pa[c], false, vb, (short)0, acc, false, false);
    }
#pragma unroll
    for (int r = 0; r < 8; ++r) {
      const int m = m0 + r + 8 * lhalf;
      const int d = dt * 16 + l15;
      O[base + (size_t)m * EDIM + d] = acc[r] * rinv[r];
    }
  }
}

extern "C" void kernel_launch(void* const* d_in, const int* in_sizes, int n_in,
                              void* d_out, int out_size, void* d_ws, size_t ws_size,
                              hipStream_t stream) {
  (void)in_sizes; (void)n_in; (void)d_ws; (void)ws_size; (void)out_size;
  const float* Q = (const float*)d_in[0];
  const float* K = (const float*)d_in[1];
  const float* V = (const float*)d_in[2];
  // d_in[3] (attn_mask) unused by the reference module.
  float* O = (float*)d_out;
  dim3 grid(BATCH * NSEQ);   // 4096 independent (b,n) attention tiles
  dim3 block(256);           // 8 wave32 per block, one WGP-resident workgroup
  attn_heads_wmma_kernel<<<grid, block, 0, stream>>>(Q, K, V, O);
}
